// SSFormer_12128987644579
// MI455X (gfx1250) — matrix operand order repaired and checked by reading, not compiled, not run on hardware
//
#include <hip/hip_runtime.h>
#include <hip/hip_bf16.h>
#include <math.h>
#include <stdint.h>

// SSFormer forward for MI455X (gfx1250, wave32, WMMA).
// NOTE: the reference's N_ITERS=2 loop has no feedback (c,e never updated),
// so both iterations produce identical results -> compute once.
//
// GEMMs: v_wmma_f32_16x16x32_f16 (f16 in, f32 accumulate).
// Attention K/V tiles staged into LDS with GLOBAL_LOAD_ASYNC_TO_LDS_B128
// (3-buffer pipeline, ASYNCcnt + one workgroup barrier per s-tile).
// Softmax / BN / swish in f32.

typedef __attribute__((ext_vector_type(16))) _Float16 v16h;
typedef __attribute__((ext_vector_type(8)))  _Float16 h8;
typedef __attribute__((ext_vector_type(8)))  float    v8f;

#define B_  8
#define C_  512
#define NH  4
#define CH  128            // channels per head
#define L_  1024           // H*W
#define BH  (B_ * NH)      // 32
#define ST  64             // s-tile (columns of logits per iteration)
#define NT_ (L_ / ST)      // 16 s-iterations
#define QK_SCALE 0.29730177875068026f   // 128^(-1/4), applied to q and k

#define S_WAIT_ASYNC(n) asm volatile("s_wait_asynccnt " #n ::: "memory")

// One lane-quad of async copy: 16 bytes global -> LDS (ASYNCcnt-tracked).
static __device__ __forceinline__ void async_ld16(uint32_t lds_byte_off,
                                                  const void* gaddr) {
  asm volatile("global_load_async_to_lds_b128 %0, %1, off"
               :: "v"(lds_byte_off), "v"(gaddr)
               : "memory");
}

static __device__ __forceinline__ uint32_t lds_off(const void* p) {
  return (uint32_t)(uintptr_t)p;   // flat addr[31:0] == LDS byte offset
}

static __device__ __forceinline__ v8f zero8() {
  v8f z;
#pragma unroll
  for (int i = 0; i < 8; ++i) z[i] = 0.0f;
  return z;
}

// A-fragment (16x32 f16, MxK): lane<16: row=lane, K {k0..k0+15};
// lane>=16: row=lane-16, K {k0+16..k0+31}.  Two 16B loads.
static __device__ __forceinline__ v16h load_afrag(const _Float16* row_k0,
                                                  bool hiHalf) {
  const _Float16* p = row_k0 + (hiHalf ? 16 : 0);
  h8 lo = *(const h8*)(p);
  h8 hi = *(const h8*)(p + 8);
  v16h r;
#pragma unroll
  for (int i = 0; i < 8; ++i) { r[i] = lo[i]; r[8 + i] = hi[i]; }
  return r;
}

// ---------------------------------------------------------------------------
// Prep: f32 [B,C,L] -> f16 [B,C,L] (V operand) and f16 [B*NH, L, CH] * scale
// ---------------------------------------------------------------------------
__global__ __launch_bounds__(256) void prep_cast_transpose(
    const float* __restrict__ in, _Float16* __restrict__ p16,
    _Float16* __restrict__ pT) {
  int idx = blockIdx.x * 256 + threadIdx.x;      // < B*C*L = 2^22
  float v = in[idx];
  p16[idx] = (_Float16)v;
  int l   = idx & (L_ - 1);
  int chn = (idx >> 10) & (C_ - 1);
  int b   = idx >> 19;                           // C*L = 2^19
  int h   = chn >> 7, cc = chn & (CH - 1);
  pT[((size_t)(b * NH + h) * L_ + l) * CH + cc] = (_Float16)(v * QK_SCALE);
}

__global__ __launch_bounds__(256) void cast_weights(
    const float* __restrict__ w, _Float16* __restrict__ w16) {
  int idx = blockIdx.x * 256 + threadIdx.x;      // < C*C
  w16[idx] = (_Float16)w[idx];
}

// ---------------------------------------------------------------------------
// Flash attention: out[t,cc] = softmax_s( q[t,:]·k[s,:] ) @ v[cc,s]
//   qT,kT : [BH, L, CH] f16 (pre-scaled), K-contig.  v: [B, C, L] f16.
// 256 threads = 8 waves; wave owns 16 query rows; block owns 128 rows of one
// (b,h).  K/V s-tiles (ST=64) staged in LDS via async copy, 3 buffers.
// ---------------------------------------------------------------------------
__global__ __launch_bounds__(256) void attn_kernel(
    const _Float16* __restrict__ qT, const _Float16* __restrict__ kT,
    const _Float16* __restrict__ v, _Float16* __restrict__ out, int ostride,
    float oscale) {
  // LDS: 3x K[ST][CH] + 3x V[CH][ST] + per-wave P[16][ST]  = 112 KB
  __shared__ _Float16 smem[6 * ST * CH + 8 * 16 * ST];
  _Float16* Kb = smem;                       // Kb + buf*ST*CH
  _Float16* Vb = smem + 3 * ST * CH;         // Vb + buf*CH*ST
  _Float16* Pb = smem + 6 * ST * CH;         // Pb + wave*16*ST

  const int tid  = threadIdx.x;
  const int lane = tid & 31;
  const int wv   = tid >> 5;
  const bool hiH = lane >= 16;
  const int ln   = lane & 15;
  const int blk  = blockIdx.x;               // BH * (L/128)
  const int bh   = blk >> 3;
  const int b    = bh >> 2;
  const int h    = bh & 3;
  const int tb   = (blk & 7) * 128 + wv * 16;

  const _Float16* qbase = qT + (size_t)bh * L_ * CH;
  const _Float16* kbase = kT + (size_t)bh * L_ * CH;
  const _Float16* vbase = v + ((size_t)b * C_ + (size_t)h * CH) * L_;

  // cooperative async copy of s-tile j into buffer j%3 (8 x b128 per thread)
  auto issue_tile = [&](int j) {
    int buf = j % 3;
    {  // K tile: ST*CH f16 = 16KB, contiguous in global
      const _Float16* g = kbase + (size_t)j * ST * CH;
      uint32_t l0 = lds_off(Kb + buf * ST * CH);
#pragma unroll
      for (int k = 0; k < 4; ++k) {
        int o8 = (tid + k * 256) * 8;        // f16 index, 16B granules
        async_ld16(l0 + o8 * 2, g + o8);
      }
    }
    {  // V tile: CH rows x ST f16 (128B per row)
      uint32_t l0 = lds_off(Vb + buf * CH * ST);
#pragma unroll
      for (int k = 0; k < 4; ++k) {
        int o = tid + k * 256;
        int cc = o >> 3, seg = (o & 7) * 8;
        async_ld16(l0 + (cc * ST + seg) * 2,
                   vbase + (size_t)cc * L_ + j * ST + seg);
      }
    }
  };

  // Q A-fragments, K-dim 0..127 (4-step WMMA chain), kept in registers
  v16h aq[4];
  {
    const _Float16* qr = qbase + (size_t)(tb + ln) * CH;
#pragma unroll
    for (int kk = 0; kk < 4; ++kk) aq[kk] = load_afrag(qr + kk * 32, hiH);
  }

  v8f acc[8];
#pragma unroll
  for (int i = 0; i < 8; ++i) acc[i] = zero8();
  float m[8], l[8];
#pragma unroll
  for (int r = 0; r < 8; ++r) { m[r] = -1e30f; l[r] = 0.0f; }

  issue_tile(0);
  issue_tile(1);

  for (int j = 0; j < NT_; ++j) {
    const int buf = j % 3;
    // all but the most recent tile's 8 ops are complete -> tile j is in LDS
    S_WAIT_ASYNC(8);
    __syncthreads();

    // ---- logits: four 16(t) x 16(s) tiles, K = 128, from LDS ----
    // Per kk step: issue all 8 ds_load_b128 as a group, then 4 WMMAs, so the
    // next group's loads overlap XDL execution.
    v8f w[4];
#pragma unroll
    for (int sub = 0; sub < 4; ++sub) w[sub] = zero8();
    {
      const _Float16* kb =
          Kb + buf * ST * CH + (size_t)ln * CH + (hiH ? 16 : 0);
#pragma unroll
      for (int kk = 0; kk < 4; ++kk) {
        v16h bk0 = *(const v16h*)(kb + 0 * 16 * CH + kk * 32);
        v16h bk1 = *(const v16h*)(kb + 1 * 16 * CH + kk * 32);
        v16h bk2 = *(const v16h*)(kb + 2 * 16 * CH + kk * 32);
        v16h bk3 = *(const v16h*)(kb + 3 * 16 * CH + kk * 32);
        w[0] = __builtin_amdgcn_wmma_f32_16x16x32_f16(
            false, aq[kk], false, bk0, (short)0, w[0], false, false);
        w[1] = __builtin_amdgcn_wmma_f32_16x16x32_f16(
            false, aq[kk], false, bk1, (short)0, w[1], false, false);
        w[2] = __builtin_amdgcn_wmma_f32_16x16x32_f16(
            false, aq[kk], false, bk2, (short)0, w[2], false, false);
        w[3] = __builtin_amdgcn_wmma_f32_16x16x32_f16(
            false, aq[kk], false, bk3, (short)0, w[3], false, false);
      }
    }

    // ---- online softmax over 64 columns; P stored to LDS as computed ----
    _Float16* pw = Pb + wv * 16 * ST;
    float corr[8];
#pragma unroll
    for (int r = 0; r < 8; ++r) {
      float t = fmaxf(fmaxf(w[0][r], w[1][r]), fmaxf(w[2][r], w[3][r]));
#pragma unroll
      for (int off = 1; off < 16; off <<= 1) t = fmaxf(t, __shfl_xor(t, off, 32));
      float mn = fmaxf(m[r], t);
      corr[r] = __expf(m[r] - mn);
      m[r] = mn;
      int row = r + (hiH ? 8 : 0);
      float s = 0.0f;
#pragma unroll
      for (int sub = 0; sub < 4; ++sub) {
        float pv = __expf(w[sub][r] - mn);
        s += pv;
        pw[row * ST + sub * 16 + ln] = (_Float16)pv;   // LDS store, latency
      }                                                // hidden under rescale
#pragma unroll
      for (int off = 1; off < 16; off <<= 1) s += __shfl_xor(s, off, 32);
      l[r] = l[r] * corr[r] + s;
    }
#pragma unroll
    for (int nt = 0; nt < 8; ++nt)
#pragma unroll
      for (int r = 0; r < 8; ++r) acc[nt][r] *= corr[r];

    // ---- P A-fragments (same-wave DS ordering guarantees visibility) ----
    v16h ap0 = load_afrag(pw + (size_t)ln * ST, hiH);
    v16h ap1 = load_afrag(pw + (size_t)ln * ST + 32, hiH);

    // ---- O += P @ V^T : 8 cc-tiles, K = 64 (s), from LDS ----
    const _Float16* vb = Vb + buf * CH * ST + (size_t)ln * ST + (hiH ? 16 : 0);
#pragma unroll
    for (int nt = 0; nt < 8; ++nt) {
      v16h b0 = *(const v16h*)(vb + nt * 16 * ST);
      v16h b1 = *(const v16h*)(vb + nt * 16 * ST + 32);
      acc[nt] = __builtin_amdgcn_wmma_f32_16x16x32_f16(
          false, ap0, false, b0, (short)0, acc[nt], false, false);
      acc[nt] = __builtin_amdgcn_wmma_f32_16x16x32_f16(
          false, ap1, false, b1, (short)0, acc[nt], false, false);
    }

    // buffer (j+2)%3 held tile j-1, consumed by all waves before barrier j
    if (j + 2 < NT_) issue_tile(j + 2);
  }

  // ---- normalize and store f16 ----
  size_t obase = (ostride == CH) ? (size_t)bh * L_ * CH
                                 : (size_t)b * L_ * C_ + (size_t)h * CH;
  _Float16* op = out + obase;
  float inv[8];
#pragma unroll
  for (int r = 0; r < 8; ++r) inv[r] = oscale / l[r];
#pragma unroll
  for (int nt = 0; nt < 8; ++nt)
#pragma unroll
    for (int r = 0; r < 8; ++r) {
      int row = tb + r + (hiH ? 8 : 0);
      op[(size_t)row * ostride + nt * 16 + ln] = (_Float16)(acc[nt][r] * inv[r]);
    }
}

// ---------------------------------------------------------------------------
// 1x1 conv as GEMM: x[o,n] = sum_c W[o,c] * eT[n,c] + bias[o],  n = b*L + l
// Block tile 128x128 (waves 4Mx2N, each wave 32x64). x written f32 to d_out.
// ---------------------------------------------------------------------------
__global__ __launch_bounds__(256) void conv_gemm(
    const _Float16* __restrict__ w16, const _Float16* __restrict__ eT,
    const float* __restrict__ bias, float* __restrict__ x) {
  const int lane = threadIdx.x & 31;
  const int wv   = threadIdx.x >> 5;
  const bool hiH = lane >= 16;
  const int ln   = lane & 15;
  const int mb   = blockIdx.x >> 6;              // 0..3
  const int nb   = blockIdx.x & 63;              // 0..63
  const int m0   = mb * 128 + (wv >> 1) * 32;
  const int n0   = nb * 128 + (wv & 1) * 64;

  v8f acc[2][4];
#pragma unroll
  for (int i = 0; i < 2; ++i)
#pragma unroll
    for (int j = 0; j < 4; ++j) acc[i][j] = zero8();

  for (int k0 = 0; k0 < C_; k0 += 32) {
    v16h a[2];
#pragma unroll
    for (int mt = 0; mt < 2; ++mt)
      a[mt] = load_afrag(w16 + (size_t)(m0 + mt * 16 + ln) * C_ + k0, hiH);
#pragma unroll
    for (int nt = 0; nt < 4; ++nt) {
      const _Float16* er =
          eT + (size_t)(n0 + nt * 16 + ln) * C_ + k0 + (hiH ? 16 : 0);
      v16h bm = *(const v16h*)er;
      acc[0][nt] = __builtin_amdgcn_wmma_f32_16x16x32_f16(
          false, a[0], false, bm, (short)0, acc[0][nt], false, false);
      acc[1][nt] = __builtin_amdgcn_wmma_f32_16x16x32_f16(
          false, a[1], false, bm, (short)0, acc[1][nt], false, false);
    }
  }

#pragma unroll
  for (int mt = 0; mt < 2; ++mt)
#pragma unroll
    for (int nt = 0; nt < 4; ++nt)
#pragma unroll
      for (int r = 0; r < 8; ++r) {
        int o = m0 + mt * 16 + r + (hiH ? 8 : 0);
        int n = n0 + nt * 16 + ln;
        int bb = n >> 10, ll = n & (L_ - 1);
        x[((size_t)bb * C_ + o) * L_ + ll] = acc[mt][nt][r] + bias[o];
      }
}

// ---------------------------------------------------------------------------
// BatchNorm training-mode stats (one block per channel) + swish apply.
// ---------------------------------------------------------------------------
__global__ __launch_bounds__(256) void bn_stats(const float* __restrict__ x,
                                                float* __restrict__ mean,
                                                float* __restrict__ var) {
  __shared__ float s1[256], s2[256];
  const int o = blockIdx.x, tid = threadIdx.x;
  float s = 0.0f, q = 0.0f;
  for (int n = tid; n < B_ * L_; n += 256) {
    int b = n >> 10, l = n & (L_ - 1);
    float v = x[((size_t)b * C_ + o) * L_ + l];
    s += v; q += v * v;
  }
  s1[tid] = s; s2[tid] = q;
  __syncthreads();
  for (int st = 128; st > 0; st >>= 1) {
    if (tid < st) { s1[tid] += s1[tid + st]; s2[tid] += s2[tid + st]; }
    __syncthreads();
  }
  if (tid == 0) {
    float mu = s1[0] * (1.0f / (B_ * L_));
    mean[o] = mu;
    var[o]  = s2[0] * (1.0f / (B_ * L_)) - mu * mu;
  }
}

__global__ __launch_bounds__(256) void bn_swish(
    float* __restrict__ x, const float* __restrict__ mean,
    const float* __restrict__ var, const float* __restrict__ gamma,
    const float* __restrict__ beta) {
  int idx = blockIdx.x * 256 + threadIdx.x;      // < B*C*L
  int o = (idx >> 10) & (C_ - 1);
  float v  = x[idx];
  float xn = (v - mean[o]) * rsqrtf(var[o] + 1e-5f) * gamma[o] + beta[o];
  x[idx] = xn * (1.0f / (1.0f + __expf(-xn)));
}

// ---------------------------------------------------------------------------
extern "C" void kernel_launch(void* const* d_in, const int* in_sizes, int n_in,
                              void* d_out, int out_size, void* d_ws,
                              size_t ws_size, hipStream_t stream) {
  (void)in_sizes; (void)n_in; (void)out_size; (void)ws_size;
  const float* c      = (const float*)d_in[0];
  const float* e      = (const float*)d_in[1];
  const float* conv_w = (const float*)d_in[2];
  const float* conv_b = (const float*)d_in[3];
  const float* gamma  = (const float*)d_in[4];
  const float* beta   = (const float*)d_in[5];
  float* out = (float*)d_out;

  const size_t NEL = (size_t)B_ * C_ * L_;       // 4,194,304
  char* ws = (char*)d_ws;
  size_t off = 0;
  auto carve = [&](size_t bytes) -> void* {
    void* p = ws + off;
    off = (off + bytes + 255) & ~(size_t)255;
    return p;
  };
  _Float16* c16   = (_Float16*)carve(NEL * 2);   // [B,C,L]
  _Float16* e16   = (_Float16*)carve(NEL * 2);   // [B,C,L]
  _Float16* cT16  = (_Float16*)carve(NEL * 2);   // [BH,L,CH] * scale
  _Float16* eT16  = (_Float16*)carve(NEL * 2);   // [BH,L,CH] * scale
  _Float16* cAT16 = (_Float16*)carve(NEL * 2);   // _c transposed, * scale
  _Float16* eAT16 = (_Float16*)carve(NEL * 2);   // _e as [B,L,C]
  _Float16* w16   = (_Float16*)carve((size_t)C_ * C_ * 2);
  float* mean     = (float*)carve(C_ * 4);
  float* var      = (float*)carve(C_ * 4);

  const dim3 blk(256);
  prep_cast_transpose<<<dim3(NEL / 256), blk, 0, stream>>>(c, c16, cT16);
  prep_cast_transpose<<<dim3(NEL / 256), blk, 0, stream>>>(e, e16, eT16);
  cast_weights<<<dim3((C_ * C_) / 256), blk, 0, stream>>>(conv_w, w16);
  // _c = attn(q=e, k=c, v=c)   (output pre-scaled: it is step B's q)
  attn_kernel<<<dim3(BH * (L_ / 128)), blk, 0, stream>>>(
      eT16, cT16, c16, cAT16, CH, QK_SCALE);
  // _e = attn(q=_c, k=e, v=e) -> [B, L, C] for the conv GEMM
  attn_kernel<<<dim3(BH * (L_ / 128)), blk, 0, stream>>>(
      cAT16, eT16, e16, eAT16, C_, 1.0f);
  // x = W @ _e + bias  (f32 into d_out)
  conv_gemm<<<dim3(256), blk, 0, stream>>>(w16, eAT16, conv_b, out);
  bn_stats<<<dim3(C_), blk, 0, stream>>>(out, mean, var);
  bn_swish<<<dim3(NEL / 256), blk, 0, stream>>>(out, mean, var, gamma, beta);
}
